// ColBert_71554155151626
// MI455X (gfx1250) — matrix operand order repaired and checked by reading, not compiled, not run on hardware
//
#include <hip/hip_runtime.h>
#include <hip/hip_bf16.h>

typedef __attribute__((ext_vector_type(16))) _Float16 v16h;
typedef __attribute__((ext_vector_type(8)))  float    v8f;

#define WMMA_F16(A, B, C) \
  __builtin_amdgcn_wmma_f32_16x16x32_f16(false, (A), false, (B), (short)0, (C), false, false)

// ---------------------------------------------------------------------------
// Phase 0: one-shot convert W [64,768] fp32 -> f16 (96 KB, L2-resident after).
// ---------------------------------------------------------------------------
__global__ __launch_bounds__(256) void convert_w_kernel(
    const float* __restrict__ W, _Float16* __restrict__ Wh)
{
  const int i = (blockIdx.x * 256 + threadIdx.x) * 4;   // 49152 / 4 = 12288 threads
#pragma unroll
  for (int e = 0; e < 4; ++e) Wh[i + e] = (_Float16)W[i + e];
}

// ---------------------------------------------------------------------------
// Phase 1: P = X[ntok,768] @ W^T, L2-normalize rows, emit f16.
// One wave handles a 16-token x 64-output tile: 4 accumulators, 24 K-steps.
// B fragments come pre-converted (f16) -> no cvt in the W path.
// ---------------------------------------------------------------------------
__global__ __launch_bounds__(256) void proj_norm_kernel(
    const float* __restrict__ X, const _Float16* __restrict__ Wh,
    _Float16* __restrict__ out, int ntok)
{
  const int lane  = threadIdx.x & 31;
  const int wave  = threadIdx.x >> 5;
  const int tile  = blockIdx.x * 8 + wave;
  const int tokBase = tile * 16;
  if (tokBase >= ntok) return;

  const int lhalf = lane >> 4;   // K-half selector
  const int lrow  = lane & 15;   // M row (A) / N col (B,C)

  const float* xrow = X + (size_t)(tokBase + lrow) * 768;

  v8f acc[4] = {v8f{}, v8f{}, v8f{}, v8f{}};

  for (int kb = 0; kb < 768; kb += 32) {
    // Prefetch next K-step of the streaming X row while WMMAs run
    __builtin_prefetch(xrow + kb + 32, 0, 0);
    // A fragment: two contiguous 8-float runs per lane, converted to f16
    v16h a;
    {
      const float* p0 = xrow + kb + 16 * lhalf;
      const float* p1 = p0 + 8;
#pragma unroll
      for (int e = 0; e < 8; ++e) {
        a[e]     = (_Float16)p0[e];
        a[8 + e] = (_Float16)p1[e];
      }
    }
    // B fragments: 16 contiguous f16 per lane from pre-converted W
#pragma unroll
    for (int nt = 0; nt < 4; ++nt) {
      const _Float16* wrow = Wh + (size_t)(nt * 16 + lrow) * 768 + kb + 16 * lhalf;
      v16h b;
#pragma unroll
      for (int e = 0; e < 16; ++e) b[e] = wrow[e];
      acc[nt] = WMMA_F16(a, b, acc[nt]);
    }
  }

  // Row-wise L2 normalize: sum of squares across 4 N-tiles + 16 lanes of a half
#pragma unroll
  for (int r = 0; r < 8; ++r) {
    float ss = acc[0][r] * acc[0][r] + acc[1][r] * acc[1][r] +
               acc[2][r] * acc[2][r] + acc[3][r] * acc[3][r];
    ss += __shfl_xor(ss, 1);
    ss += __shfl_xor(ss, 2);
    ss += __shfl_xor(ss, 4);
    ss += __shfl_xor(ss, 8);
    const float scale = 1.0f / fmaxf(sqrtf(ss), 1e-12f);
    const int row = tokBase + r + 8 * lhalf;
    _Float16* orow = out + (size_t)row * 64;
#pragma unroll
    for (int nt = 0; nt < 4; ++nt)
      orow[lrow + 16 * nt] = (_Float16)(acc[nt][r] * scale);
  }
}

// ---------------------------------------------------------------------------
// Phase 2: MaxSim. One wave per (qb, db) pair: sim[32 x 180] via WMMA (K=64),
// running max over doc tokens, lane-reduced, summed over 32 query tokens.
// 8 waves in a block share qb -> q fragments hit WGP$ / L2.
// ---------------------------------------------------------------------------
__global__ __launch_bounds__(256) void maxsim_kernel(
    const _Float16* __restrict__ qn, const _Float16* __restrict__ dn,
    float* __restrict__ scores)
{
  const int lane = threadIdx.x & 31;
  const int wave = threadIdx.x >> 5;
  const int pair = blockIdx.x * 8 + wave;   // 0..8191
  const int qb   = pair >> 8;               // 0..31
  const int db   = pair & 255;              // 0..255

  const int lhalf = lane >> 4;
  const int lrow  = lane & 15;

  // Preload 4 A fragments: 2 M-tiles x 2 K-steps of the 32x64 query block
  v16h a[2][2];
#pragma unroll
  for (int mt = 0; mt < 2; ++mt) {
    const _Float16* qrow = qn + (size_t)(qb * 32 + mt * 16 + lrow) * 64;
#pragma unroll
    for (int s = 0; s < 2; ++s) {
      const _Float16* p0 = qrow + 32 * s + 16 * lhalf;
      const _Float16* p1 = p0 + 8;
#pragma unroll
      for (int e = 0; e < 8; ++e) {
        a[mt][s][e]     = p0[e];
        a[mt][s][8 + e] = p1[e];
      }
    }
  }

  float rmax0[8], rmax1[8];
#pragma unroll
  for (int r = 0; r < 8; ++r) { rmax0[r] = -3.0e38f; rmax1[r] = -3.0e38f; }

  const _Float16* dbase = dn + (size_t)db * 180 * 64;

  for (int nt = 0; nt < 12; ++nt) {           // 192 padded doc tokens
    const int j = nt * 16 + lrow;             // this lane's doc-token column
    const bool valid = (j < 180);
    v16h b0 = {}, b1 = {};
    if (valid) {
      const _Float16* drow = dbase + (size_t)j * 64;
      const _Float16* p0 = drow + 16 * lhalf;        // K-step 0
      const _Float16* p1 = drow + 32 + 16 * lhalf;   // K-step 1
#pragma unroll
      for (int e = 0; e < 16; ++e) { b0[e] = p0[e]; b1[e] = p1[e]; }
    }
    v8f c0 = {}, c1 = {};
    c0 = WMMA_F16(a[0][0], b0, c0);
    c0 = WMMA_F16(a[0][1], b1, c0);
    c1 = WMMA_F16(a[1][0], b0, c1);
    c1 = WMMA_F16(a[1][1], b1, c1);
    if (valid) {
#pragma unroll
      for (int r = 0; r < 8; ++r) {
        rmax0[r] = fmaxf(rmax0[r], c0[r]);
        rmax1[r] = fmaxf(rmax1[r], c1[r]);
      }
    }
  }

  // Per-row max across the 16 lanes of each half, then sum the 32 row-maxes
  float s = 0.0f;
#pragma unroll
  for (int r = 0; r < 8; ++r) {
    float m0 = rmax0[r], m1 = rmax1[r];
    m0 = fmaxf(m0, __shfl_xor(m0, 1)); m1 = fmaxf(m1, __shfl_xor(m1, 1));
    m0 = fmaxf(m0, __shfl_xor(m0, 2)); m1 = fmaxf(m1, __shfl_xor(m1, 2));
    m0 = fmaxf(m0, __shfl_xor(m0, 4)); m1 = fmaxf(m1, __shfl_xor(m1, 4));
    m0 = fmaxf(m0, __shfl_xor(m0, 8)); m1 = fmaxf(m1, __shfl_xor(m1, 8));
    s += m0 + m1;                        // rows (mt, r, half=lane>>4)
  }
  s += __shfl_xor(s, 16);                // combine the two halves
  if (lane == 0) scores[qb * 256 + db] = s;
}

// ---------------------------------------------------------------------------
// Phase 3: cross-entropy over scores[32, 256], labels = q * group_size.
// ---------------------------------------------------------------------------
__global__ __launch_bounds__(256) void ce_loss_kernel(
    const float* __restrict__ scores, const int* __restrict__ gsp,
    float* __restrict__ out)
{
  __shared__ float red[256];
  const int t = threadIdx.x;
  const int g = gsp[0];
  float lsum = 0.0f;

  for (int q = 0; q < 32; ++q) {
    const float v = scores[q * 256 + t];
    red[t] = v; __syncthreads();
    for (int off = 128; off > 0; off >>= 1) {
      if (t < off) red[t] = fmaxf(red[t], red[t + off]);
      __syncthreads();
    }
    const float m = red[0]; __syncthreads();
    red[t] = expf(v - m); __syncthreads();
    for (int off = 128; off > 0; off >>= 1) {
      if (t < off) red[t] += red[t + off];
      __syncthreads();
    }
    const float lse = m + logf(red[0]); __syncthreads();
    if (t == q * g) lsum += lse - v;
  }

  red[t] = lsum; __syncthreads();
  for (int off = 128; off > 0; off >>= 1) {
    if (t < off) red[t] += red[t + off];
    __syncthreads();
  }
  if (t == 0) out[0] = red[0] / 32.0f;
}

// ---------------------------------------------------------------------------
extern "C" void kernel_launch(void* const* d_in, const int* in_sizes, int n_in,
                              void* d_out, int out_size, void* d_ws, size_t ws_size,
                              hipStream_t stream) {
  (void)in_sizes; (void)n_in; (void)out_size; (void)ws_size;
  const float* qry = (const float*)d_in[0];   // [32, 32, 768]
  const float* doc = (const float*)d_in[1];   // [256, 180, 768]
  const float* W   = (const float*)d_in[2];   // [64, 768]
  const int*   gs  = (const int*)d_in[3];     // group_size
  float* out = (float*)d_out;

  char* ws = (char*)d_ws;
  _Float16* Wh     = (_Float16*)ws;                                  // 64*768 f16
  _Float16* q_norm = (_Float16*)(ws + (size_t)64 * 768 * 2);         // 1024*64 f16
  _Float16* d_norm = (_Float16*)(ws + (size_t)64 * 768 * 2
                                    + (size_t)1024 * 64 * 2);        // 46080*64 f16
  float*    scores = (float*)(ws + (size_t)64 * 768 * 2
                                 + (size_t)1024 * 64 * 2
                                 + (size_t)46080 * 64 * 2);          // 32*256 f32

  // W fp32 -> f16 once (49152 elems, 4 per thread -> 48 blocks)
  convert_w_kernel<<<48, 256, 0, stream>>>(W, Wh);
  // 1024 q-tokens = 64 tiles -> 8 blocks; 46080 d-tokens = 2880 tiles -> 360 blocks
  proj_norm_kernel<<<8,    256, 0, stream>>>(qry, Wh, q_norm, 1024);
  proj_norm_kernel<<<360,  256, 0, stream>>>(doc, Wh, d_norm, 46080);
  // 32*256 = 8192 pairs, 8 waves per block
  maxsim_kernel<<<1024, 256, 0, stream>>>(q_norm, d_norm, scores);
  ce_loss_kernel<<<1,   256, 0, stream>>>(scores, gs, out);
}
